// xlstm_45303315038910
// MI455X (gfx1250) — compile-verified, matrix operand order checked by reading, not run on hardware
//
#include <hip/hip_runtime.h>
#include <math.h>

// ---------------- model constants ----------------
#define Bsz    256
#define Lsz    512
#define ENCsz  128
#define TGTsz  96
#define N2sz   256
#define Dsz    256
#define INNERsz 512
#define MNHsz  2
#define MDHsz  256
#define SNHsz  4
#define SDHsz  64
#define Kconv  4
#define MAKsz  25
#define FFsz   384
#define Ssz    128                 // sequence length inside blocks (= ENC)
#define NTOK   (Bsz * ENCsz)       // 32768 tokens

typedef __bf16 bf16;
typedef __attribute__((ext_vector_type(16))) __bf16 bf16x16;
typedef __attribute__((ext_vector_type(8)))  __bf16 bf16x8;
typedef __attribute__((ext_vector_type(8)))  float  f32x8;

// LDS byte-offset of a __shared__ object (generic -> addrspace(3) -> 32-bit offset)
__device__ __forceinline__ unsigned lds_addr_of(const void* p) {
  return (unsigned)(unsigned long long)(__attribute__((address_space(3))) const void*)p;
}

#define S_WAIT_ASYNC(N) asm volatile("s_wait_asynccnt " #N ::: "memory")

// =====================================================================
// Generic batched bf16 WMMA GEMM:  C = alpha * (A @ B^T) + bias + resid
//   A: (M x K) row-major, lda ; B: (N x K) row-major, ldb (weights as (N,K))
//   C: (M x N) row-major, ldc, fp32
// Block: 256 threads = 8 waves, tile 128(M) x 64(N); each wave 32x32.
// A/B K-slabs (128x32 / 64x32 bf16) are staged into LDS with
// global_load_async_to_lds_b128 (ASYNCcnt), double-buffered, so stage i+1's
// DMA overlaps stage i's WMMA; fragments then come from LDS (ds_load_b128).
// Contract (guaranteed by the host code): M is a multiple of 128; the B
// allocation is readable up to ceil(N/64)*64 rows (pad weight copies);
// K multiple of 32; lda/ldb multiples of 8.  This keeps the issue path
// branch-free: every thread issues exactly 3 async loads per stage, so
// s_wait_asynccnt 3 precisely retires the previous slab.
// =====================================================================
__device__ __forceinline__ void async_stage(const bf16* __restrict__ A, const bf16* __restrict__ B,
                                            int lda, int ldb, int mblk, int nblk, int k0,
                                            unsigned ldsA, unsigned ldsB, int tid) {
  const int row = tid >> 2, seg = tid & 3;
  // A tile: 128 rows x 32 bf16 = 512 x 16B chunks, 2 per thread (rows r, r+64)
  {
    unsigned long long gp0 = (unsigned long long)(A + (long)(mblk + row) * lda + k0 + seg * 8);
    unsigned long long gp1 = (unsigned long long)(A + (long)(mblk + row + 64) * lda + k0 + seg * 8);
    unsigned lp0 = ldsA + (unsigned)tid * 16u;
    unsigned lp1 = ldsA + (unsigned)(tid + 256) * 16u;
    asm volatile("global_load_async_to_lds_b128 %0, %1, off" :: "v"(lp0), "v"(gp0) : "memory");
    asm volatile("global_load_async_to_lds_b128 %0, %1, off" :: "v"(lp1), "v"(gp1) : "memory");
  }
  // B tile: 64 rows x 32 bf16 = 256 x 16B chunks, 1 per thread
  {
    unsigned long long gp = (unsigned long long)(B + (long)(nblk + row) * ldb + k0 + seg * 8);
    unsigned lp = ldsB + (unsigned)tid * 16u;
    asm volatile("global_load_async_to_lds_b128 %0, %1, off" :: "v"(lp), "v"(gp) : "memory");
  }
}

// fragment from an LDS tile laid out row-major, 32 bf16 per row
__device__ __forceinline__ bf16x16 frag_lds(const bf16* __restrict__ base, int row0, int lane) {
  int half = lane >> 4;                 // 0: K 0..7 & 16..23 ; 1: K 8..15 & 24..31
  int r = row0 + (lane & 15);
  const bf16* rp = base + r * 32 + half * 8;
  bf16x8 lo = *(const bf16x8*)(rp);
  bf16x8 hi = *(const bf16x8*)(rp + 16);
  bf16x16 f;
#pragma unroll
  for (int i = 0; i < 8; ++i) { f[i] = lo[i]; f[i + 8] = hi[i]; }
  return f;
}

__global__ __launch_bounds__(256) void wmma_gemm_bf16(
    const bf16* __restrict__ A, const bf16* __restrict__ B, float* __restrict__ C,
    const float* __restrict__ bias, const float* __restrict__ resid,
    int M, int N, int K, int lda, int ldb, int ldc,
    long batchA, long batchB, long batchC, float alpha) {
  __shared__ bf16 As[2 * 128 * 32];
  __shared__ bf16 Bs[2 * 64 * 32];

  const int bz = blockIdx.z;
  A += (long)bz * batchA;
  B += (long)bz * batchB;
  C += (long)bz * batchC;
  const float* Rp = resid ? resid + (long)bz * batchC : nullptr;

  const int tid  = threadIdx.x;
  const int lane = tid & 31;
  const int wave = tid >> 5;
  const int mblk = blockIdx.y * 128;
  const int nblk = blockIdx.x * 64;
  const int wmrow = (wave & 3) * 32;   // wave M sub-tile inside block tile
  const int wnrow = (wave >> 2) * 32;  // wave N sub-tile inside block tile
  const int m0 = mblk + wmrow;
  const int n0 = nblk + wnrow;

  const unsigned ldsA0 = lds_addr_of(As);
  const unsigned ldsB0 = lds_addr_of(Bs);

  f32x8 acc00 = {0,0,0,0,0,0,0,0};
  f32x8 acc01 = {0,0,0,0,0,0,0,0};
  f32x8 acc10 = {0,0,0,0,0,0,0,0};
  f32x8 acc11 = {0,0,0,0,0,0,0,0};

  const int nsteps = K >> 5;
  async_stage(A, B, lda, ldb, mblk, nblk, 0, ldsA0, ldsB0, tid);

  for (int i = 0; i < nsteps; ++i) {
    const int buf = i & 1;
    if (i + 1 < nsteps) {
      // kick off next slab into the other buffer, then wait for current slab only
      async_stage(A, B, lda, ldb, mblk, nblk, (i + 1) * 32,
                  ldsA0 + (unsigned)((buf ^ 1) * 128 * 32 * 2),
                  ldsB0 + (unsigned)((buf ^ 1) * 64 * 32 * 2), tid);
      S_WAIT_ASYNC(3);                 // 3 in-flight = next slab; current slab done
    } else {
      S_WAIT_ASYNC(0);
    }
    __syncthreads();

    const bf16* Ab = As + buf * (128 * 32);
    const bf16* Bb = Bs + buf * (64 * 32);
    bf16x16 a0 = frag_lds(Ab, wmrow,      lane);
    bf16x16 a1 = frag_lds(Ab, wmrow + 16, lane);
    bf16x16 b0 = frag_lds(Bb, wnrow,      lane);
    bf16x16 b1 = frag_lds(Bb, wnrow + 16, lane);
    acc00 = __builtin_amdgcn_wmma_f32_16x16x32_bf16(false, a0, false, b0, (short)0, acc00, false, false);
    acc01 = __builtin_amdgcn_wmma_f32_16x16x32_bf16(false, a0, false, b1, (short)0, acc01, false, false);
    acc10 = __builtin_amdgcn_wmma_f32_16x16x32_bf16(false, a1, false, b0, (short)0, acc10, false, false);
    acc11 = __builtin_amdgcn_wmma_f32_16x16x32_bf16(false, a1, false, b1, (short)0, acc11, false, false);
    __syncthreads();                   // all waves done reading buf before its next overwrite
  }

  const int cr = (lane >> 4) * 8;      // row sub-offset inside 16x16 tile
  const int cc = lane & 15;            // col inside tile
#define STORE_TILE(ACC, RM, RN)                                           \
  do {                                                                    \
    int rm = (RM), rn = (RN);                                             \
    if (rm < M && rn < N) {                                               \
      int col = rn + cc;                                                  \
      float bv = bias ? bias[col] : 0.f;                                  \
      _Pragma("unroll") for (int v = 0; v < 8; ++v) {                     \
        long idx = (long)(rm + cr + v) * ldc + col;                       \
        float val = ACC[v] * alpha + bv;                                  \
        if (Rp) val += Rp[idx];                                           \
        C[idx] = val;                                                     \
      }                                                                   \
    }                                                                     \
  } while (0)
  STORE_TILE(acc00, m0,      n0);
  STORE_TILE(acc01, m0,      n0 + 16);
  STORE_TILE(acc10, m0 + 16, n0);
  STORE_TILE(acc11, m0 + 16, n0 + 16);
#undef STORE_TILE
}

// =====================================================================
// Stage A: moving-average decomposition -> [res | ma] bf16 rows (B*ENC x 1024)
// =====================================================================
__global__ void moving_avg_kernel(const float* __restrict__ x, bf16* __restrict__ rmcat) {
  long idx = (long)blockIdx.x * blockDim.x + threadIdx.x;
  const long total = (long)Bsz * Lsz * ENCsz;
  if (idx >= total) return;
  int e = (int)(idx % ENCsz);
  int l = (int)((idx / ENCsz) % Lsz);
  int b = (int)(idx / ((long)ENCsz * Lsz));
  const float* xb = x + (long)b * Lsz * ENCsz + e;
  float s = 0.f;
#pragma unroll
  for (int j = 0; j < MAKsz; ++j) {
    int t = l + j - (MAKsz - 1) / 2;
    t = t < 0 ? 0 : (t > Lsz - 1 ? Lsz - 1 : t);   // edge padding
    s += xb[(long)t * ENCsz];
  }
  float ma = s * (1.f / MAKsz);
  float xv = xb[(long)l * ENCsz];
  long row = (long)b * ENCsz + e;
  rmcat[row * 1024 + l]       = (bf16)(xv - ma);
  rmcat[row * 1024 + 512 + l] = (bf16)ma;
}

__global__ void pack_wcat_kernel(const float* __restrict__ Ws, const float* __restrict__ Wt,
                                 const float* __restrict__ bs, const float* __restrict__ bt,
                                 bf16* __restrict__ Wcat, float* __restrict__ bias_st) {
  int idx = blockIdx.x * blockDim.x + threadIdx.x;
  if (idx < TGTsz * Lsz) {
    int n = idx / Lsz, l = idx % Lsz;
    Wcat[(long)n * 1024 + l]       = (bf16)Ws[idx];
    Wcat[(long)n * 1024 + 512 + l] = (bf16)Wt[idx];
  }
  if (idx < TGTsz) bias_st[idx] = bs[idx] + bt[idx];
}

__global__ void cvt_bf16_kernel(const float* __restrict__ in, bf16* __restrict__ out, long n) {
  long i = (long)blockIdx.x * blockDim.x + threadIdx.x;
  if (i < n) out[i] = (bf16)in[i];
}

// ---------------- batchnorm over (B, N2) per ENC channel ----------------
__global__ __launch_bounds__(256) void bn_reduce_kernel(const float* __restrict__ h,
                                                        float* __restrict__ mu, float* __restrict__ rstd) {
  int e = blockIdx.x, tid = threadIdx.x;
  float s1 = 0.f, s2 = 0.f;
  for (int i = tid; i < Bsz * N2sz; i += 256) {
    int b = i / N2sz, n = i % N2sz;
    float v = h[((long)b * ENCsz + e) * N2sz + n];
    s1 += v; s2 += v * v;
  }
  __shared__ float l1[256], l2[256];
  l1[tid] = s1; l2[tid] = s2; __syncthreads();
  for (int off = 128; off > 0; off >>= 1) {
    if (tid < off) { l1[tid] += l1[tid + off]; l2[tid] += l2[tid + off]; }
    __syncthreads();
  }
  if (tid == 0) {
    float m = l1[0] / (float)(Bsz * N2sz);
    float v = l2[0] / (float)(Bsz * N2sz) - m * m;
    mu[e] = m; rstd[e] = rsqrtf(v + 1e-5f);
  }
}

__global__ void bn_apply_kernel(float* __restrict__ h, const float* __restrict__ mu,
                                const float* __restrict__ rstd, const float* __restrict__ bnw,
                                const float* __restrict__ bnb) {
  long i = (long)blockIdx.x * blockDim.x + threadIdx.x;
  const long total = (long)NTOK * N2sz;
  if (i >= total) return;
  int e = (int)((i / N2sz) % ENCsz);
  h[i] = (h[i] - mu[e]) * rstd[e] * bnw[e] + bnb[e];
}

// ---------------- row layernorm (rowlen 256), emits bf16 (+opt fp32) ----------------
__global__ __launch_bounds__(256) void ln256_kernel(const float* __restrict__ x, const float* __restrict__ w,
                                                    bf16* __restrict__ out_bf, float* __restrict__ out_f) {
  long row = blockIdx.x; int tid = threadIdx.x;
  float v = x[row * 256 + tid];
  __shared__ float l1[256], l2[256];
  l1[tid] = v; l2[tid] = v * v; __syncthreads();
  for (int off = 128; off > 0; off >>= 1) {
    if (tid < off) { l1[tid] += l1[tid + off]; l2[tid] += l2[tid + off]; }
    __syncthreads();
  }
  float mu  = l1[0] * (1.f / 256.f);
  float var = l2[0] * (1.f / 256.f) - mu * mu;
  float y = (v - mu) * rsqrtf(var + 1e-5f) * w[tid];
  out_bf[row * 256 + tid] = (bf16)y;
  if (out_f) out_f[row * 256 + tid] = y;
}

// ---------------- causal depthwise conv (K=4, zero pad left) + silu ----------------
__global__ void conv_silu_kernel(const float* __restrict__ in, int in_ld, int C,
                                 const float* __restrict__ w, const float* __restrict__ bconv,
                                 float* __restrict__ out_f, bf16* __restrict__ out_bf) {
  long i = (long)blockIdx.x * blockDim.x + threadIdx.x;
  const long total = (long)NTOK * C;
  if (i >= total) return;
  int c = (int)(i % C);
  long t = i / C;
  int s = (int)(t & (Ssz - 1));
  long brow = t - s;
  float acc = bconv[c];
#pragma unroll
  for (int k = 0; k < Kconv; ++k) {
    int sp = s - (Kconv - 1) + k;
    if (sp >= 0) acc += w[c * Kconv + k] * in[(brow + sp) * in_ld + c];
  }
  float y = acc / (1.f + expf(-acc));
  out_f[t * C + c] = y;
  if (out_bf) out_bf[t * C + c] = (bf16)y;
}

// ---------------- mLSTM: headwise 4x4 q/k/v projections ----------------
__global__ void mlstm_qkv_kernel(const float* __restrict__ up, const float* __restrict__ xc,
                                 const float* __restrict__ Wq, const float* __restrict__ Wk,
                                 const float* __restrict__ Wv,
                                 bf16* __restrict__ qh, bf16* __restrict__ kh, bf16* __restrict__ vT) {
  long i = (long)blockIdx.x * blockDim.x + threadIdx.x;
  const long total = (long)NTOK * 512;
  if (i >= total) return;
  int c = (int)(i % 512);
  long t = i / 512;
  int hd = c >> 2, e = c & 3;
  const float* wq = Wq + hd * 16 + e * 4;
  const float* wk = Wk + hd * 16 + e * 4;
  const float* wv = Wv + hd * 16 + e * 4;
  float q = 0.f, k = 0.f, v = 0.f;
#pragma unroll
  for (int d = 0; d < 4; ++d) {
    float xcv = xc[t * 512 + hd * 4 + d];
    float xiv = up[t * 1024 + hd * 4 + d];     // x_in = first half of up
    q += wq[d] * xcv;
    k += wk[d] * xcv;
    v += wv[d] * xiv;
  }
  int s = (int)(t & (Ssz - 1));
  long b = t >> 7;
  int h2 = c >> 8, dd = c & 255;
  long bh = b * MNHsz + h2;
  qh[bh * (Ssz * MDHsz) + (long)s * MDHsz + dd] = (bf16)q;
  kh[bh * (Ssz * MDHsz) + (long)s * MDHsz + dd] = (bf16)k;
  vT[bh * (MDHsz * Ssz) + (long)dd * Ssz + s]   = (bf16)v;
}

// ---------------- mLSTM: i/f gate logits = qkv . Wig/Wfg ----------------
__global__ __launch_bounds__(256) void mlstm_gates_kernel(
    const bf16* __restrict__ qh, const bf16* __restrict__ kh, const bf16* __restrict__ vT,
    const float* __restrict__ Wig, const float* __restrict__ big,
    const float* __restrict__ Wfg, const float* __restrict__ bfg,
    float* __restrict__ i_pre, float* __restrict__ f_pre) {
  long t = blockIdx.x; int tid = threadIdx.x;
  int s = (int)(t & (Ssz - 1)); long b = t >> 7;
  float acc0 = 0.f, acc1 = 0.f, acc2 = 0.f, acc3 = 0.f;
  for (int j = tid; j < 3 * INNERsz; j += 256) {
    int seg = j >> 9;              // 0:q 1:k 2:v
    int dloc = j & 511;
    int h2 = dloc >> 8, dd = dloc & 255;
    long bh = b * MNHsz + h2;
    float val;
    if (seg == 0)      val = (float)qh[bh * (Ssz * MDHsz) + (long)s * MDHsz + dd];
    else if (seg == 1) val = (float)kh[bh * (Ssz * MDHsz) + (long)s * MDHsz + dd];
    else               val = (float)vT[bh * (MDHsz * Ssz) + (long)dd * Ssz + s];
    acc0 += val * Wig[j];
    acc1 += val * Wig[3 * INNERsz + j];
    acc2 += val * Wfg[j];
    acc3 += val * Wfg[3 * INNERsz + j];
  }
  __shared__ float r0[256], r1[256], r2[256], r3[256];
  r0[tid] = acc0; r1[tid] = acc1; r2[tid] = acc2; r3[tid] = acc3;
  __syncthreads();
  for (int off = 128; off > 0; off >>= 1) {
    if (tid < off) {
      r0[tid] += r0[tid + off]; r1[tid] += r1[tid + off];
      r2[tid] += r2[tid + off]; r3[tid] += r3[tid + off];
    }
    __syncthreads();
  }
  if (tid == 0) {
    i_pre[(b * 2 + 0) * Ssz + s] = r0[0] + big[0];
    i_pre[(b * 2 + 1) * Ssz + s] = r1[0] + big[1];
    f_pre[(b * 2 + 0) * Ssz + s] = r2[0] + bfg[0];
    f_pre[(b * 2 + 1) * Ssz + s] = r3[0] + bfg[1];
  }
}

__device__ __forceinline__ float log_sigmoid_f(float x) {
  return fminf(x, 0.f) - log1pf(expf(-fabsf(x)));
}

// inclusive cumsum of log_sigmoid(f_pre) along S per (b, head)
__global__ __launch_bounds__(128) void logsig_cumsum_kernel(const float* __restrict__ f_pre,
                                                            float* __restrict__ fc) {
  int row = blockIdx.x, tid = threadIdx.x;
  float v = log_sigmoid_f(f_pre[(long)row * Ssz + tid]);
  __shared__ float buf[Ssz];
  buf[tid] = v; __syncthreads();
  for (int off = 1; off < Ssz; off <<= 1) {
    float t = (tid >= off) ? buf[tid - off] : 0.f;
    __syncthreads();
    buf[tid] += t;
    __syncthreads();
  }
  fc[(long)row * Ssz + tid] = buf[tid];
}

// masked log-decay gating + row normalization between the two attention GEMMs
__global__ __launch_bounds__(128) void mlstm_attn_ew_kernel(
    const float* __restrict__ scores, const float* __restrict__ fc,
    const float* __restrict__ i_pre, bf16* __restrict__ CmA) {
  int i = blockIdx.x;        // query row
  int bh = blockIdx.y;       // batch*MNH + head
  int j = threadIdx.x;
  long base = (long)bh * (Ssz * Ssz);
  float fi = fc[(long)bh * Ssz + i];
  bool valid = (j <= i);
  float logD = valid ? (fi - fc[(long)bh * Ssz + j] + i_pre[(long)bh * Ssz + j]) : -3.4e38f;
  __shared__ float red[Ssz];
  red[j] = logD; __syncthreads();
  for (int off = 64; off > 0; off >>= 1) {
    if (j < off) red[j] = fmaxf(red[j], red[j + off]);
    __syncthreads();
  }
  float maxD = red[0];
  __syncthreads();
  float Dm = valid ? expf(logD - maxD) : 0.f;
  float Cm = scores[base + (long)i * Ssz + j] * Dm;   // scores already scaled by 1/sqrt(MDH)
  red[j] = Cm; __syncthreads();
  for (int off = 64; off > 0; off >>= 1) {
    if (j < off) red[j] += red[j + off];
    __syncthreads();
  }
  float norm = fmaxf(fabsf(red[0]), expf(-maxD));
  CmA[base + (long)i * Ssz + j] = (bf16)(Cm / (norm + 1e-6f));
}

// per-head norm of attention output + *onw + skip*xc, * silu(z) -> bf16 A for Wdown
__global__ __launch_bounds__(512) void mlstm_post_kernel(
    const float* __restrict__ att, const float* __restrict__ xc, const float* __restrict__ up,
    const float* __restrict__ onw, const float* __restrict__ skip, bf16* __restrict__ Adown) {
  long t = blockIdx.x; int tid = threadIdx.x;      // tid = d in [0,512)
  int h2 = tid >> 8, dd = tid & 255;
  int s = (int)(t & (Ssz - 1)); long b = t >> 7;
  float hv = att[(((b * MNHsz + h2) * Ssz) + s) * (long)MDHsz + dd];
  __shared__ float l1[512], l2[512];
  l1[tid] = hv; l2[tid] = hv * hv; __syncthreads();
  for (int off = 128; off > 0; off >>= 1) {
    if ((tid & 255) < off) { l1[tid] += l1[tid + off]; l2[tid] += l2[tid + off]; }
    __syncthreads();
  }
  int segbase = h2 << 8;
  float mu  = l1[segbase] * (1.f / 256.f);
  float var = l2[segbase] * (1.f / 256.f) - mu * mu;
  float y = (hv - mu) * rsqrtf(var + 1e-5f) * onw[tid];
  float xcv = xc[t * 512 + tid];
  float z = up[t * 1024 + 512 + tid];
  float silz = z / (1.f + expf(-z));
  Adown[t * 512 + tid] = (bf16)((y + skip[tid] * xcv) * silz);
}

// ---------------- sLSTM sequential scan: one block per (batch, head) ----------------
__global__ __launch_bounds__(256) void slstm_scan_kernel(
    const float* __restrict__ gi, const float* __restrict__ gf,
    const float* __restrict__ gz, const float* __restrict__ go,
    const float* __restrict__ R, const float* __restrict__ rb,
    float* __restrict__ hs) {
  int blk = blockIdx.x;
  int head = blk & (SNHsz - 1);
  long b = blk >> 2;
  int tid = threadIdx.x;
  int g = tid >> 6, e = tid & 63;
  __shared__ float hprev[64], cst[64], nst[64], mst[64], raw[256];
  if (tid < 64) { hprev[tid] = 0.f; cst[tid] = 0.f; nst[tid] = 0.f; mst[tid] = 0.f; }
  __syncthreads();
  const float* wxp = (g == 0) ? gi : (g == 1) ? gf : (g == 2) ? gz : go;
  const float* Rh = R + (long)head * SDHsz * 4 * SDHsz;
  float rbv = rb[(head * 4 + g) * SDHsz + e];
  for (int s = 0; s < Ssz; ++s) {
    long t = b * Ssz + s;
    float acc = wxp[t * 256 + head * SDHsz + e] + rbv;
    for (int d = 0; d < SDHsz; ++d)
      acc += hprev[d] * Rh[(d * 4 + g) * SDHsz + e];
    raw[tid] = acc;
    __syncthreads();
    if (tid < 64) {
      float ir = raw[e], fr = raw[64 + e], zr = raw[128 + e], orr = raw[192 + e];
      float lfm = mst[e] + log_sigmoid_f(fr);
      float mn = fmaxf(ir, lfm);
      float ig = expf(ir - mn);
      float fg = expf(lfm - mn);
      float cn = fg * cst[e] + ig * tanhf(zr);
      float nn = fg * nst[e] + ig;
      float hn = (1.f / (1.f + expf(-orr))) * cn / nn;
      cst[e] = cn; nst[e] = nn; mst[e] = mn; hprev[e] = hn;
      hs[t * 256 + head * SDHsz + e] = hn;
    }
    __syncthreads();
  }
}

// per-head (64) norm of sLSTM output * gnw, residual add into h
__global__ __launch_bounds__(256) void slstm_post_kernel(const float* __restrict__ hs,
                                                         const float* __restrict__ gnw,
                                                         float* __restrict__ h) {
  long t = blockIdx.x; int tid = threadIdx.x;
  float v = hs[t * 256 + tid];
  __shared__ float l1[256], l2[256];
  l1[tid] = v; l2[tid] = v * v; __syncthreads();
  for (int off = 32; off > 0; off >>= 1) {
    if ((tid & 63) < off) { l1[tid] += l1[tid + off]; l2[tid] += l2[tid + off]; }
    __syncthreads();
  }
  int base = tid & ~63;
  float mu  = l1[base] * (1.f / 64.f);
  float var = l2[base] * (1.f / 64.f) - mu * mu;
  h[t * 256 + tid] += (v - mu) * rsqrtf(var + 1e-5f) * gnw[tid];
}

// ---------------- FFN gating: gelu(g) * h -> bf16 ----------------
__global__ void ffn_glu_kernel(const float* __restrict__ up2, bf16* __restrict__ a) {
  long i = (long)blockIdx.x * blockDim.x + threadIdx.x;
  const long total = (long)NTOK * FFsz;
  if (i >= total) return;
  int j = (int)(i % FFsz);
  long t = i / FFsz;
  float g  = up2[t * (2 * FFsz) + j];
  float hh = up2[t * (2 * FFsz) + FFsz + j];
  float ge = 0.5f * g * (1.f + erff(g * 0.7071067811865476f));
  a[t * FFsz + j] = (bf16)(ge * hh);
}

// ---------------- final (B,S,TGT) -> (B,TGT,S) transpose ----------------
__global__ void out_transpose_kernel(const float* __restrict__ y, float* __restrict__ out) {
  long i = (long)blockIdx.x * blockDim.x + threadIdx.x;
  const long total = (long)Bsz * ENCsz * TGTsz;
  if (i >= total) return;
  int n = (int)(i % TGTsz);
  long be = i / TGTsz;
  int e = (int)(be & (ENCsz - 1));
  long b = be >> 7;
  out[(b * TGTsz + n) * ENCsz + e] = y[i];
}

// =====================================================================
// Host side
// =====================================================================
static inline long cdiv(long a, long b) { return (a + b - 1) / b; }

static void gemm(hipStream_t st, const bf16* A, const bf16* B, float* C,
                 const float* bias, const float* resid,
                 int M, int N, int K, int lda, int ldb, int ldc,
                 long bA, long bB, long bC, int batches, float alpha) {
  dim3 grid((unsigned)cdiv(N, 64), (unsigned)cdiv(M, 128), (unsigned)batches);
  wmma_gemm_bf16<<<grid, 256, 0, st>>>(A, B, C, bias, resid, M, N, K, lda, ldb, ldc, bA, bB, bC, alpha);
}

static void cvt(hipStream_t st, const float* src, bf16* dst, long n) {
  cvt_bf16_kernel<<<(unsigned)cdiv(n, 256), 256, 0, st>>>(src, dst, n);
}

struct MlstmScratch {
  bf16* xln_bf; float* up; float* xc;
  bf16 *qh, *kh, *vT;
  float *i_pre, *f_pre, *fc;
  float* scores; bf16* CmA; float* att; bf16* Adown;
};

static void run_mlstm(hipStream_t s, float* h,
                      const float* ln_w, const bf16* Wup_bf,
                      const float* conv_w, const float* conv_b,
                      const float* Wq, const float* Wk, const float* Wv,
                      const float* Wig, const float* big, const float* Wfg, const float* bfg,
                      const float* onw, const float* skip, const bf16* Wdown_bf,
                      const MlstmScratch& t) {
  ln256_kernel<<<NTOK, 256, 0, s>>>(h, ln_w, t.xln_bf, nullptr);
  gemm(s, t.xln_bf, Wup_bf, t.up, nullptr, nullptr, NTOK, 2 * INNERsz, Dsz, Dsz, Dsz, 2 * INNERsz, 0, 0, 0, 1, 1.f);
  {
    long n = (long)NTOK * INNERsz;
    conv_silu_kernel<<<(unsigned)cdiv(n, 256), 256, 0, s>>>(t.up, 2 * INNERsz, INNERsz, conv_w, conv_b, t.xc, nullptr);
    mlstm_qkv_kernel<<<(unsigned)cdiv(n, 256), 256, 0, s>>>(t.up, t.xc, Wq, Wk, Wv, t.qh, t.kh, t.vT);
  }
  mlstm_gates_kernel<<<NTOK, 256, 0, s>>>(t.qh, t.kh, t.vT, Wig, big, Wfg, bfg, t.i_pre, t.f_pre);
  logsig_cumsum_kernel<<<Bsz * MNHsz, Ssz, 0, s>>>(t.f_pre, t.fc);
  // scores = (q @ k^T) / sqrt(MDH)  — batched over B*MNH
  gemm(s, t.qh, t.kh, t.scores, nullptr, nullptr,
       Ssz, Ssz, MDHsz, MDHsz, MDHsz, Ssz,
       (long)Ssz * MDHsz, (long)Ssz * MDHsz, (long)Ssz * Ssz, Bsz * MNHsz, 1.f / 16.f);
  {
    dim3 g(Ssz, Bsz * MNHsz);
    mlstm_attn_ew_kernel<<<g, Ssz, 0, s>>>(t.scores, t.fc, t.i_pre, t.CmA);
  }
  // att = Cm @ v  — B stored as v^T (MDH x S)
  gemm(s, t.CmA, t.vT, t.att, nullptr, nullptr,
       Ssz, MDHsz, Ssz, Ssz, Ssz, MDHsz,
       (long)Ssz * Ssz, (long)MDHsz * Ssz, (long)Ssz * MDHsz, Bsz * MNHsz, 1.f);
  mlstm_post_kernel<<<NTOK, 512, 0, s>>>(t.att, t.xc, t.up, onw, skip, t.Adown);
  // h = h + Adown @ Wdown^T   (residual folded into GEMM epilogue)
  gemm(s, t.Adown, Wdown_bf, h, nullptr, h, NTOK, Dsz, INNERsz, INNERsz, INNERsz, Dsz, 0, 0, 0, 1, 1.f);
}

// input index map (setup_inputs dict order, depth-first)
enum {
  IN_X = 0, IN_WSEAS, IN_BSEAS, IN_WTREND, IN_BTREND, IN_WMM, IN_BMM, IN_BNW, IN_BNB,
  IN_WMM2, IN_BMM2, IN_POSTLN,
  IN_B0_LN, IN_B0_WUP, IN_B0_CONVW, IN_B0_CONVB, IN_B0_WQ, IN_B0_WK, IN_B0_WV,
  IN_B0_WIG, IN_B0_BIG, IN_B0_WFG, IN_B0_BFG, IN_B0_ONW, IN_B0_SKIP, IN_B0_WDOWN,
  IN_B1_LN1, IN_B1_CONVW, IN_B1_CONVB, IN_B1_WI, IN_B1_WF, IN_B1_WZ, IN_B1_WO,
  IN_B1_R, IN_B1_RB, IN_B1_GNW, IN_B1_LN2, IN_B1_WFFUP, IN_B1_WFFDOWN,
  IN_B2_LN, IN_B2_WUP, IN_B2_CONVW, IN_B2_CONVB, IN_B2_WQ, IN_B2_WK, IN_B2_WV,
  IN_B2_WIG, IN_B2_BIG, IN_B2_WFG, IN_B2_BFG, IN_B2_ONW, IN_B2_SKIP, IN_B2_WDOWN
};

extern "C" void kernel_launch(void* const* d_in, const int* in_sizes, int n_in,
                              void* d_out, int out_size, void* d_ws, size_t ws_size,
                              hipStream_t stream) {
  (void)in_sizes; (void)n_in; (void)out_size; (void)ws_size;
  auto F = [&](int i) { return (const float*)d_in[i]; };

  // ---------------- workspace layout (deterministic bump allocator) ----------------
  char* base = (char*)d_ws;
  size_t off = 0;
  auto alloc = [&](size_t bytes) -> char* {
    char* p = base + off;
    off += (bytes + 255) & ~(size_t)255;
    return p;
  };

  // persistent region.  NOTE: B-operand weight copies whose N is not a
  // multiple of 64 (Wcat, Wmm2: N=96) are padded to 128 rows so the async
  // tile DMA can read full 64-row tiles without clamping.
  float* h       = (float*)alloc((size_t)NTOK * Dsz * 4);
  float* mu      = (float*)alloc(ENCsz * 4);
  float* rstd    = (float*)alloc(ENCsz * 4);
  float* bias_st = (float*)alloc(TGTsz * 4);
  bf16* Wcat_bf  = (bf16*)alloc((size_t)128 * 1024 * 2);       // 96 used, 128 padded
  bf16* Wmm_bf   = (bf16*)alloc((size_t)N2sz * TGTsz * 2);
  bf16* Wmm2_bf  = (bf16*)alloc((size_t)128 * Dsz * 2);        // 96 used, 128 padded
  bf16* Wup0_bf  = (bf16*)alloc((size_t)2 * INNERsz * Dsz * 2);
  bf16* Wdn0_bf  = (bf16*)alloc((size_t)Dsz * INNERsz * 2);
  bf16* Wup2_bf  = (bf16*)alloc((size_t)2 * INNERsz * Dsz * 2);
  bf16* Wdn2_bf  = (bf16*)alloc((size_t)Dsz * INNERsz * 2);
  bf16* Wi_bf    = (bf16*)alloc((size_t)SNHsz * SDHsz * SDHsz * 2);
  bf16* Wf_bf    = (bf16*)alloc((size_t)SNHsz * SDHsz * SDHsz * 2);
  bf16* Wz_bf    = (bf16*)alloc((size_t)SNHsz * SDHsz * SDHsz * 2);
  bf16* Wo_bf    = (bf16*)alloc((size_t)SNHsz * SDHsz * SDHsz * 2);
  bf16* Wfu_bf   = (bf16*)alloc((size_t)2 * FFsz * Dsz * 2);
  bf16* Wfd_bf   = (bf16*)alloc((size_t)Dsz * FFsz * 2);
  size_t scratch0 = off;

  // weight conversions
  cvt(stream, F(IN_WMM), Wmm_bf, (long)N2sz * TGTsz);
  cvt(stream, F(IN_WMM2), Wmm2_bf, (long)TGTsz * Dsz);
  cvt(stream, F(IN_B0_WUP), Wup0_bf, (long)2 * INNERsz * Dsz);
  cvt(stream, F(IN_B0_WDOWN), Wdn0_bf, (long)Dsz * INNERsz);
  cvt(stream, F(IN_B2_WUP), Wup2_bf, (long)2 * INNERsz * Dsz);
  cvt(stream, F(IN_B2_WDOWN), Wdn2_bf, (long)Dsz * INNERsz);
  cvt(stream, F(IN_B1_WI), Wi_bf, (long)SNHsz * SDHsz * SDHsz);
  cvt(stream, F(IN_B1_WF), Wf_bf, (long)SNHsz * SDHsz * SDHsz);
  cvt(stream, F(IN_B1_WZ), Wz_bf, (long)SNHsz * SDHsz * SDHsz);
  cvt(stream, F(IN_B1_WO), Wo_bf, (long)SNHsz * SDHsz * SDHsz);
  cvt(stream, F(IN_B1_WFFUP), Wfu_bf, (long)2 * FFsz * Dsz);
  cvt(stream, F(IN_B1_WFFDOWN), Wfd_bf, (long)Dsz * FFsz);
  pack_wcat_kernel<<<(unsigned)cdiv((long)TGTsz * Lsz, 256), 256, 0, stream>>>(
      F(IN_WSEAS), F(IN_WTREND), F(IN_BSEAS), F(IN_BTREND), Wcat_bf, bias_st);

  // ---------------- Stage A: decomposition + projections + batchnorm ----------------
  {
    off = scratch0;
    bf16*  rmcat = (bf16*)alloc((size_t)NTOK * 1024 * 2);
    float* stbuf = (float*)alloc((size_t)NTOK * TGTsz * 4);
    bf16*  st_bf = (bf16*)alloc((size_t)NTOK * TGTsz * 2);

    moving_avg_kernel<<<(unsigned)cdiv((long)Bsz * Lsz * ENCsz, 256), 256, 0, stream>>>(F(IN_X), rmcat);
    // st = res @ Wseas^T + ma @ Wtrend^T + (bseas+btrend)   (K packed to 1024)
    gemm(stream, rmcat, Wcat_bf, stbuf, bias_st, nullptr,
         NTOK, TGTsz, 1024, 1024, 1024, TGTsz, 0, 0, 0, 1, 1.f);
    cvt(stream, stbuf, st_bf, (long)NTOK * TGTsz);
    // h = st @ Wmm^T + bmm
    gemm(stream, st_bf, Wmm_bf, h, F(IN_BMM), nullptr,
         NTOK, N2sz, TGTsz, TGTsz, TGTsz, N2sz, 0, 0, 0, 1, 1.f);
    bn_reduce_kernel<<<ENCsz, 256, 0, stream>>>(h, mu, rstd);
    bn_apply_kernel<<<(unsigned)cdiv((long)NTOK * N2sz, 256), 256, 0, stream>>>(h, mu, rstd, F(IN_BNW), F(IN_BNB));
  }

  // ---------------- mLSTM scratch (shared by block0 and block2) ----------------
  off = scratch0;
  MlstmScratch mt;
  mt.xln_bf = (bf16*)alloc((size_t)NTOK * Dsz * 2);
  mt.up     = (float*)alloc((size_t)NTOK * 2 * INNERsz * 4);
  mt.xc     = (float*)alloc((size_t)NTOK * INNERsz * 4);
  mt.qh     = (bf16*)alloc((size_t)NTOK * INNERsz * 2);
  mt.kh     = (bf16*)alloc((size_t)NTOK * INNERsz * 2);
  mt.vT     = (bf16*)alloc((size_t)NTOK * INNERsz * 2);
  mt.i_pre  = (float*)alloc((size_t)Bsz * MNHsz * Ssz * 4);
  mt.f_pre  = (float*)alloc((size_t)Bsz * MNHsz * Ssz * 4);
  mt.fc     = (float*)alloc((size_t)Bsz * MNHsz * Ssz * 4);
  mt.scores = (float*)alloc((size_t)Bsz * MNHsz * Ssz * Ssz * 4);
  mt.CmA    = (bf16*)alloc((size_t)Bsz * MNHsz * Ssz * Ssz * 2);
  mt.att    = (float*)alloc((size_t)Bsz * MNHsz * Ssz * MDHsz * 4);
  mt.Adown  = (bf16*)alloc((size_t)NTOK * INNERsz * 2);

  // ---------------- block 0: mLSTM ----------------
  run_mlstm(stream, h, F(IN_B0_LN), Wup0_bf, F(IN_B0_CONVW), F(IN_B0_CONVB),
            F(IN_B0_WQ), F(IN_B0_WK), F(IN_B0_WV), F(IN_B0_WIG), F(IN_B0_BIG),
            F(IN_B0_WFG), F(IN_B0_BFG), F(IN_B0_ONW), F(IN_B0_SKIP), Wdn0_bf, mt);

  // ---------------- block 1: sLSTM + FFN (reuse scratch region) ----------------
  {
    off = scratch0;
    bf16*  xs_bf  = (bf16*)alloc((size_t)NTOK * Dsz * 2);
    float* xs_f   = (float*)alloc((size_t)NTOK * Dsz * 4);
    float* xcs_f  = (float*)alloc((size_t)NTOK * Dsz * 4);
    bf16*  xcs_bf = (bf16*)alloc((size_t)NTOK * Dsz * 2);
    float* gi     = (float*)alloc((size_t)NTOK * Dsz * 4);
    float* gf     = (float*)alloc((size_t)NTOK * Dsz * 4);
    float* gz     = (float*)alloc((size_t)NTOK * Dsz * 4);
    float* go     = (float*)alloc((size_t)NTOK * Dsz * 4);
    float* hsbuf  = (float*)alloc((size_t)NTOK * Dsz * 4);

    ln256_kernel<<<NTOK, 256, 0, stream>>>(h, F(IN_B1_LN1), xs_bf, xs_f);
    conv_silu_kernel<<<(unsigned)cdiv((long)NTOK * Dsz, 256), 256, 0, stream>>>(
        xs_f, Dsz, Dsz, F(IN_B1_CONVW), F(IN_B1_CONVB), xcs_f, xcs_bf);
    // block-diagonal headwise projections: batched over 4 heads (A col offset 64)
    gemm(stream, xcs_bf, Wi_bf, gi, nullptr, nullptr, NTOK, SDHsz, SDHsz, Dsz, SDHsz, Dsz,
         SDHsz, (long)SDHsz * SDHsz, SDHsz, SNHsz, 1.f);
    gemm(stream, xcs_bf, Wf_bf, gf, nullptr, nullptr, NTOK, SDHsz, SDHsz, Dsz, SDHsz, Dsz,
         SDHsz, (long)SDHsz * SDHsz, SDHsz, SNHsz, 1.f);
    gemm(stream, xs_bf, Wz_bf, gz, nullptr, nullptr, NTOK, SDHsz, SDHsz, Dsz, SDHsz, Dsz,
         SDHsz, (long)SDHsz * SDHsz, SDHsz, SNHsz, 1.f);
    gemm(stream, xs_bf, Wo_bf, go, nullptr, nullptr, NTOK, SDHsz, SDHsz, Dsz, SDHsz, Dsz,
         SDHsz, (long)SDHsz * SDHsz, SDHsz, SNHsz, 1.f);
    slstm_scan_kernel<<<Bsz * SNHsz, 256, 0, stream>>>(gi, gf, gz, go, F(IN_B1_R), F(IN_B1_RB), hsbuf);
    slstm_post_kernel<<<NTOK, 256, 0, stream>>>(hsbuf, F(IN_B1_GNW), h);

    // FFN
    bf16*  xf_bf = (bf16*)alloc((size_t)NTOK * Dsz * 2);
    float* up2   = (float*)alloc((size_t)NTOK * 2 * FFsz * 4);
    bf16*  a_bf  = (bf16*)alloc((size_t)NTOK * FFsz * 2);
    ln256_kernel<<<NTOK, 256, 0, stream>>>(h, F(IN_B1_LN2), xf_bf, nullptr);
    gemm(stream, xf_bf, Wfu_bf, up2, nullptr, nullptr,
         NTOK, 2 * FFsz, Dsz, Dsz, Dsz, 2 * FFsz, 0, 0, 0, 1, 1.f);
    ffn_glu_kernel<<<(unsigned)cdiv((long)NTOK * FFsz, 256), 256, 0, stream>>>(up2, a_bf);
    gemm(stream, a_bf, Wfd_bf, h, nullptr, h,
         NTOK, Dsz, FFsz, FFsz, FFsz, Dsz, 0, 0, 0, 1, 1.f);
  }

  // ---------------- block 2: mLSTM ----------------
  run_mlstm(stream, h, F(IN_B2_LN), Wup2_bf, F(IN_B2_CONVW), F(IN_B2_CONVB),
            F(IN_B2_WQ), F(IN_B2_WK), F(IN_B2_WV), F(IN_B2_WIG), F(IN_B2_BIG),
            F(IN_B2_WFG), F(IN_B2_BFG), F(IN_B2_ONW), F(IN_B2_SKIP), Wdn2_bf, mt);

  // ---------------- final: post_ln -> Wmm2 -> transpose ----------------
  {
    off = scratch0;
    bf16*  xp_bf = (bf16*)alloc((size_t)NTOK * Dsz * 2);
    float* yfin  = (float*)alloc((size_t)NTOK * TGTsz * 4);
    ln256_kernel<<<NTOK, 256, 0, stream>>>(h, F(IN_POSTLN), xp_bf, nullptr);
    gemm(stream, xp_bf, Wmm2_bf, yfin, F(IN_BMM2), nullptr,
         NTOK, TGTsz, Dsz, Dsz, Dsz, TGTsz, 0, 0, 0, 1, 1.f);
    out_transpose_kernel<<<(unsigned)cdiv((long)NTOK * TGTsz, 256), 256, 0, stream>>>(yfin, (float*)d_out);
  }
}
